// GroupQueryAttention_52939766890701
// MI455X (gfx1250) — matrix operand order repaired and checked by reading, not compiled, not run on hardware
//
#include <hip/hip_runtime.h>
#include <hip/hip_bf16.h>

// ---------------------------------------------------------------------------
// GQA forward for MI455X (gfx1250, wave32, WMMA).
// Qproj/KVproj (bf16 WMMA, 32x64 wave tiles) -> flash attention (bf16 WMMA,
// online softmax, 64-wide S chunks, async LDS staging) -> Oproj (-> f32).
// ---------------------------------------------------------------------------

typedef __attribute__((ext_vector_type(16))) __bf16 v16bf;
typedef __attribute__((ext_vector_type(8)))  float  v8f;
typedef __attribute__((ext_vector_type(8)))  __bf16 bf16x8;
typedef __attribute__((ext_vector_type(4)))  int    v4i;

constexpr int Bb = 2, Tt = 2048, Ss = 2048, Ee = 1024, Hh = 16, HDd = 64;
constexpr float SCALE = 0.125f; // 1/sqrt(64)

#if defined(__has_builtin)
#if __has_builtin(__builtin_amdgcn_global_load_async_to_lds_b128)
#define HAVE_ASYNC_LDS 1
#endif
#endif
#ifndef HAVE_ASYNC_LDS
#define HAVE_ASYNC_LDS 0
#endif

// Per-lane 16B global -> LDS copy (ASYNCcnt path when available).
__device__ inline void copy16_g2lds(__bf16* lds_dst, const __bf16* gsrc) {
#if HAVE_ASYNC_LDS
    __builtin_amdgcn_global_load_async_to_lds_b128(
        (__attribute__((address_space(1))) v4i*)gsrc,
        (__attribute__((address_space(3))) v4i*)lds_dst, 0, 0);
#else
    *(bf16x8*)lds_dst = *(const bf16x8*)gsrc;
#endif
}

__device__ inline void wait_async_lds() {
#if HAVE_ASYNC_LDS
#if __has_builtin(__builtin_amdgcn_s_wait_asynccnt)
    __builtin_amdgcn_s_wait_asynccnt(0);
#else
    asm volatile("s_wait_asynccnt 0x0" ::: "memory");
#endif
#endif
}

// --------------------------- WMMA helpers ----------------------------------

__device__ inline v8f wmma_bf16(v16bf a, v16bf b, v8f c) {
    return __builtin_amdgcn_wmma_f32_16x16x32_bf16(false, a, false, b,
                                                   (short)0, c, false, false);
}

// A-fragment 16x32 bf16 from row-major source (contiguous per lane: 2x b128).
__device__ inline v16bf load_a_frag(const __bf16* base, int ld) {
    const int lane = threadIdx.x & 31;
    const int m  = lane & 15;
    const int hi = lane >> 4;
    v16bf a;
#pragma unroll
    for (int v = 0; v < 4; ++v) {
        int k = hi * 8 + 2 * v;
        a[2 * v]     = base[m * ld + k];
        a[2 * v + 1] = base[m * ld + k + 1];
    }
#pragma unroll
    for (int v = 4; v < 8; ++v) {
        int k = 16 + hi * 8 + 2 * (v - 4);
        a[2 * v]     = base[m * ld + k];
        a[2 * v + 1] = base[m * ld + k + 1];
    }
    return a;
}

// B-fragment 32x16 bf16 from an N-major (transposed) source:
// B[k][n] = base[n*ld + k].  Two contiguous 16B runs per lane -> 2x ds_load_b128.
__device__ inline v16bf load_b_frag_tr(const __bf16* base, int ld) {
    const int lane = threadIdx.x & 31;
    const int n  = lane & 15;
    const int hi = lane >> 4;
    v16bf b;
#pragma unroll
    for (int v = 0; v < 8; ++v) {
        int k = hi * 8 + 2 * v + (v >= 4 ? 8 : 0);
        b[2 * v]     = base[n * ld + k];
        b[2 * v + 1] = base[n * ld + k + 1];
    }
    return b;
}

// --------------------------- GEMM + bias -----------------------------------
// C[M,N] = A[M,K] @ W[K,N] + bias[N].  256 threads = 8 waves (4M x 2N),
// wave tile 32x64 (8 WMMA / k-step), WG tile 128x128, BK = 32.

template <typename InT, typename OutT>
__global__ __launch_bounds__(256)
void gemm_bias_kernel(const InT* __restrict__ A, const float* __restrict__ W,
                      const float* __restrict__ bias, OutT* __restrict__ C,
                      int M, int N, int K) {
    __shared__ __align__(16) __bf16 As[128][40];   // 128 m x 32 k (+pad)
    __shared__ __align__(16) __bf16 Wt[128][40];   // 128 n x 32 k (transposed)

    const int tid  = threadIdx.x;
    const int wave = tid >> 5;
    const int lane = tid & 31;
    const int wm = wave >> 1;   // 0..3
    const int wn = wave & 1;    // 0..1
    const int row0 = blockIdx.x * 128;
    const int col0 = blockIdx.y * 128;

    v8f zero = {};
    v8f acc[8] = {zero, zero, zero, zero, zero, zero, zero, zero};

    for (int k0 = 0; k0 < K; k0 += 32) {
        __syncthreads();
        // Stage A block (128x32) -> bf16 LDS (coalesced global reads).
#pragma unroll
        for (int i = 0; i < 16; ++i) {
            int idx = tid + i * 256;             // 0..4095
            int r = idx >> 5, c = idx & 31;
            As[r][c] = (__bf16)(float)A[(size_t)(row0 + r) * K + (k0 + c)];
        }
        // Stage W block (32x128) transposed -> Wt[n][k].
#pragma unroll
        for (int i = 0; i < 16; ++i) {
            int idx = tid + i * 256;             // 0..4095
            int kk = idx >> 7, c = idx & 127;
            Wt[c][kk] = (__bf16)W[(size_t)(k0 + kk) * N + (col0 + c)];
        }
        if (k0 + 32 < K) {
            __builtin_prefetch(&A[(size_t)(row0 + (tid >> 1)) * K + k0 + 32], 0, 1);
            __builtin_prefetch(&W[(size_t)(k0 + 32 + (tid >> 3)) * N + col0 + (tid & 7) * 16], 0, 1);
        }
        __syncthreads();

        v16bf a0 = load_a_frag(&As[wm * 32][0], 40);
        v16bf a1 = load_a_frag(&As[wm * 32 + 16][0], 40);
#pragma unroll
        for (int j = 0; j < 4; ++j) {
            v16bf bj = load_b_frag_tr(&Wt[wn * 64 + j * 16][0], 40);
            acc[2 * j]     = wmma_bf16(a0, bj, acc[2 * j]);
            acc[2 * j + 1] = wmma_bf16(a1, bj, acc[2 * j + 1]);
        }
    }

    // Epilogue: C tile layout — VGPR r holds row (r + 8*hi), col = lane&15.
    const int n  = lane & 15;
    const int hi = lane >> 4;
#pragma unroll
    for (int j = 0; j < 4; ++j) {
#pragma unroll
        for (int im = 0; im < 2; ++im) {
            int m0 = row0 + wm * 32 + im * 16;
            int c  = col0 + wn * 64 + j * 16 + n;
            float bc = bias[c];
#pragma unroll
            for (int r = 0; r < 8; ++r) {
                float v = acc[2 * j + im][r] + bc;
                C[(size_t)(m0 + r + 8 * hi) * N + c] = (OutT)v;
            }
        }
    }
}

// --------------------------- Flash attention -------------------------------
// 256 threads = 8 waves; wave w owns T rows [t0, t0+16) for one (b, h).
// S streamed in 64-wide chunks; K staged row-major (async->LDS), V staged
// transposed so P@V B-fragments are contiguous.

__global__ __launch_bounds__(256)
void gqa_flash_kernel(const __bf16* __restrict__ Q,
                      const __bf16* __restrict__ KV,
                      __bf16* __restrict__ O) {
    __shared__ __align__(16) __bf16 Ks[64][72];      // 64 s x 64 d
    __shared__ __align__(16) __bf16 Vt[64][72];      // 64 d x 64 s (transposed)
    __shared__ __align__(16) __bf16 Ps[8][16][72];   // per-wave P tile 16x64

    const int tid  = threadIdx.x;
    const int wid  = tid >> 5;
    const int lane = tid & 31;
    const int n    = lane & 15;
    const int hi   = lane >> 4;
    const int h = blockIdx.y;
    const int b = blockIdx.z;
    const int t0 = blockIdx.x * 128 + wid * 16;

    // Q fragments (16 rows x HD=64 -> two K=32 A-fragments), loaded once.
    const __bf16* qbase = Q + ((size_t)(b * Tt + t0) * Ee + h * HDd);
    v16bf q0 = load_a_frag(qbase, Ee);
    v16bf q1 = load_a_frag(qbase + 32, Ee);

    v8f zero = {};
    v8f accO[4] = {zero, zero, zero, zero};          // 16 x 64 out tile
    float mrow[8], lrow[8];
#pragma unroll
    for (int r = 0; r < 8; ++r) { mrow[r] = -1e30f; lrow[r] = 0.f; }

    const size_t kvRow = (size_t)2 * Ee;             // 2048 elems per KV row

    for (int s0 = 0; s0 < Ss; s0 += 64) {
        __syncthreads();
        // K chunk 64x64: per-lane async 16B copies, row-major.
#pragma unroll
        for (int i = 0; i < 2; ++i) {
            int idx = tid + i * 256;                 // 0..511
            int r = idx >> 3;
            int g = (idx & 7) * 8;
            size_t off = (size_t)(b * Ss + s0 + r) * kvRow + h * HDd + g;
            copy16_g2lds(&Ks[r][g], KV + off);
        }
        // V chunk 64x64 staged transposed: Vt[d][s].
#pragma unroll
        for (int i = 0; i < 16; ++i) {
            int idx = tid + i * 256;                 // 0..4095
            int srel = idx >> 6, d = idx & 63;
            Vt[d][srel] =
                KV[(size_t)(b * Ss + s0 + srel) * kvRow + Ee + h * HDd + d];
        }
        wait_async_lds();
        __syncthreads();

        // scores = q @ k^T : 16 x 64 chunk -> 4 accumulators, 8 WMMA.
        v8f sc[4] = {zero, zero, zero, zero};
#pragma unroll
        for (int j = 0; j < 4; ++j) {
            sc[j] = wmma_bf16(q0, load_b_frag_tr(&Ks[j * 16][0],  72), sc[j]);
            sc[j] = wmma_bf16(q1, load_b_frag_tr(&Ks[j * 16][32], 72), sc[j]);
        }

        // Online softmax (rows r+8*hi live across 16-lane halves).
#pragma unroll
        for (int r = 0; r < 8; ++r) {
            float x0 = sc[0][r] * SCALE;
            float x1 = sc[1][r] * SCALE;
            float x2 = sc[2][r] * SCALE;
            float x3 = sc[3][r] * SCALE;
            float cm = fmaxf(fmaxf(x0, x1), fmaxf(x2, x3));
#pragma unroll
            for (int msk = 1; msk < 16; msk <<= 1)
                cm = fmaxf(cm, __shfl_xor(cm, msk, 32));
            float mnew  = fmaxf(mrow[r], cm);
            float alpha = __expf(mrow[r] - mnew);
            float p0 = __expf(x0 - mnew);
            float p1 = __expf(x1 - mnew);
            float p2 = __expf(x2 - mnew);
            float p3 = __expf(x3 - mnew);
            float ps = (p0 + p1) + (p2 + p3);
#pragma unroll
            for (int msk = 1; msk < 16; msk <<= 1)
                ps += __shfl_xor(ps, msk, 32);
            lrow[r] = lrow[r] * alpha + ps;
            mrow[r] = mnew;
#pragma unroll
            for (int j = 0; j < 4; ++j) accO[j][r] *= alpha;
            int m = r + 8 * hi;
            Ps[wid][m][n]      = (__bf16)p0;
            Ps[wid][m][n + 16] = (__bf16)p1;
            Ps[wid][m][n + 32] = (__bf16)p2;
            Ps[wid][m][n + 48] = (__bf16)p3;
        }
        __builtin_amdgcn_wave_barrier();

        // out += P @ V : two K=32 steps x 4 HD subtiles -> 8 WMMA.
        v16bf ap0 = load_a_frag(&Ps[wid][0][0],  72);
        v16bf ap1 = load_a_frag(&Ps[wid][0][32], 72);
#pragma unroll
        for (int j = 0; j < 4; ++j) {
            accO[j] = wmma_bf16(ap0, load_b_frag_tr(&Vt[j * 16][0],  72), accO[j]);
            accO[j] = wmma_bf16(ap1, load_b_frag_tr(&Vt[j * 16][32], 72), accO[j]);
        }
    }

    // Normalize and write attention output (bf16, [B,T,E] with head offset).
#pragma unroll
    for (int r = 0; r < 8; ++r) {
        float inv = 1.0f / lrow[r];
        int m = r + 8 * hi;
        size_t orow = (size_t)(b * Tt + t0 + m) * Ee + h * HDd + n;
#pragma unroll
        for (int j = 0; j < 4; ++j)
            O[orow + j * 16] = (__bf16)(accO[j][r] * inv);
    }
}

// --------------------------- Launcher --------------------------------------

extern "C" void kernel_launch(void* const* d_in, const int* in_sizes, int n_in,
                              void* d_out, int out_size, void* d_ws, size_t ws_size,
                              hipStream_t stream) {
    (void)in_sizes; (void)n_in; (void)out_size; (void)ws_size;

    const float* query = (const float*)d_in[0];
    const float* key   = (const float*)d_in[1];
    // d_in[2] (value) is unused: reference applies kv_proj to `key` only.
    const float* Wq  = (const float*)d_in[3];
    const float* bq  = (const float*)d_in[4];
    const float* Wkv = (const float*)d_in[5];
    const float* bkv = (const float*)d_in[6];
    const float* Wo  = (const float*)d_in[7];
    const float* bo  = (const float*)d_in[8];
    float* out = (float*)d_out;

    // Workspace: bf16 Q (8MB), bf16 KV (16MB), bf16 attn-out (8MB) = 32MB.
    __bf16* Qb  = (__bf16*)d_ws;
    __bf16* KVb = Qb  + (size_t)Bb * Tt * Ee;
    __bf16* Ab  = KVb + (size_t)Bb * Ss * 2 * Ee;

    dim3 blk(256);
    // q = query @ Wq + bq        [4096,1024] x [1024,1024]
    gemm_bias_kernel<float, __bf16>
        <<<dim3((Bb * Tt) / 128, Ee / 128), blk, 0, stream>>>(
            query, Wq, bq, Qb, Bb * Tt, Ee, Ee);
    // kv = key @ Wkv + bkv       [4096,1024] x [1024,2048]
    gemm_bias_kernel<float, __bf16>
        <<<dim3((Bb * Ss) / 128, (2 * Ee) / 128), blk, 0, stream>>>(
            key, Wkv, bkv, KVb, Bb * Ss, 2 * Ee, Ee);
    // attention
    gqa_flash_kernel<<<dim3(Tt / 128, Hh, Bb), dim3(256), 0, stream>>>(Qb, KVb, Ab);
    // out = attn @ Wo + bo       [4096,1024] x [1024,1024] -> f32
    gemm_bias_kernel<__bf16, float>
        <<<dim3((Bb * Tt) / 128, Ee / 128), blk, 0, stream>>>(
            Ab, Wo, bo, out, Bb * Tt, Ee, Ee);
}
